// Mamba_20057497272769
// MI455X (gfx1250) — matrix operand order repaired and checked by reading, not compiled, not run on hardware
//
#include <hip/hip_runtime.h>

// ---------------------------------------------------------------------------
// Mamba block forward for MI455X (gfx1250, wave32, WMMA).
// B=4, L=1024, D_MODEL=512, D_INNER=1024, D_STATE=16, DT_RANK=32, D_CONV=4
// ---------------------------------------------------------------------------

#define BATCH   4
#define SEQ     1024
#define DMODEL  512
#define DINNER  1024
#define DSTATE  16
#define DTRANK  32
#define DCONV   4
#define NTOK    (BATCH * SEQ)          // 4096 tokens

typedef __attribute__((ext_vector_type(16))) __bf16      v16bf;
typedef __attribute__((ext_vector_type(8)))  float       v8f;
typedef __attribute__((ext_vector_type(8)))  unsigned int v8u;

__device__ __forceinline__ float bf2f(unsigned short u) {
  unsigned int x = ((unsigned int)u) << 16;
  float f;
  __builtin_memcpy(&f, &x, 4);
  return f;
}
__device__ __forceinline__ unsigned short f2bf(float f) {
  unsigned int x;
  __builtin_memcpy(&x, &f, 4);
  unsigned int r = x + 0x7fffu + ((x >> 16) & 1u);   // round-to-nearest-even
  return (unsigned short)(r >> 16);
}

// ---------------------------------------------------------------------------
// f32 -> bf16 conversion
// ---------------------------------------------------------------------------
__global__ __launch_bounds__(256) void cvt_f32_bf16(const float* __restrict__ in,
                                                    unsigned short* __restrict__ out,
                                                    int n) {
  int i = blockIdx.x * 256 + threadIdx.x;
  if (i < n) out[i] = f2bf(in[i]);
}

// ---------------------------------------------------------------------------
// GEMM: C[m,n] = sum_k A[m,k] * W[n,k]   (A bf16 [M,lda], W bf16 [N,ldw])
// One wave32 per 16x64 C strip: 4 accumulators, K stepped by 32.
// Fragments are built directly from global memory (no LDS round-trip):
//   A frag, lane(r,h): row m0+r, element offsets {k0+8h .. +7} and
//                      {k0+16+8h .. +7}            (2 x b128)
//   B frag, lane(r,h): row n0+r, element offsets {k0+8h .. +7} and
//                      {k0+16+8h .. +7}            (2 x b128)
// ---------------------------------------------------------------------------
__global__ __launch_bounds__(256) void gemm_bf16_wmma(
    const unsigned short* __restrict__ A, int lda,
    const unsigned short* __restrict__ W, int ldw,
    void* __restrict__ Cout, int ldc,
    int M, int N, int K, int c_is_bf16) {
  const int tid  = threadIdx.x;
  const int wave = tid >> 5;
  const int lane = tid & 31;
  const int tiles_n = N >> 6;                // 64-wide strips
  const int tiles   = (M >> 4) * tiles_n;
  const int tile    = blockIdx.x * 8 + wave;
  if (tile >= tiles) return;                 // wave-uniform

  const int m0 = (tile / tiles_n) << 4;
  const int n0 = (tile % tiles_n) << 6;
  const int r  = lane & 15;
  const int h  = lane >> 4;

  const unsigned short* arow = A + (size_t)(m0 + r) * lda;
  const unsigned short* wrow = W + (size_t)(n0 + r) * ldw;   // +16*t*ldw per subtile

  v8f c0 = {}, c1 = {}, c2 = {}, c3 = {};

  for (int k0 = 0; k0 < K; k0 += 32) {
    if (k0 + 32 < K) {                       // next-iteration prefetch
      __builtin_prefetch(arow + k0 + 32, 0, 0);
      __builtin_prefetch(wrow + k0 + 32, 0, 0);
    }
    // A fragment (shared across the 4 subtiles)
    const uint4* pa0 = (const uint4*)(arow + k0 + h * 8);
    const uint4* pa1 = (const uint4*)(arow + k0 + 16 + h * 8);
    uint4 a0 = *pa0, a1 = *pa1;
    v8u au = {a0.x, a0.y, a0.z, a0.w, a1.x, a1.y, a1.z, a1.w};
    v16bf av = __builtin_bit_cast(v16bf, au);

    // 4 B fragments, then 4 back-to-back WMMAs
    v16bf bv[4];
#pragma unroll
    for (int t = 0; t < 4; ++t) {
      const uint4* pb = (const uint4*)(wrow + (size_t)t * 16 * ldw + k0 + h * 8);
      uint4 b0 = pb[0], b1 = pb[2];
      v8u bu = {b0.x, b0.y, b0.z, b0.w, b1.x, b1.y, b1.z, b1.w};
      bv[t] = __builtin_bit_cast(v16bf, bu);
    }
    c0 = __builtin_amdgcn_wmma_f32_16x16x32_bf16(false, av, false, bv[0], (short)0, c0, false, false);
    c1 = __builtin_amdgcn_wmma_f32_16x16x32_bf16(false, av, false, bv[1], (short)0, c1, false, false);
    c2 = __builtin_amdgcn_wmma_f32_16x16x32_bf16(false, av, false, bv[2], (short)0, c2, false, false);
    c3 = __builtin_amdgcn_wmma_f32_16x16x32_bf16(false, av, false, bv[3], (short)0, c3, false, false);
  }

  // C/D layout: lane half h, col = n-base + r, rows m0 + v + 8h
  v8f acc[4] = {c0, c1, c2, c3};
  if (c_is_bf16) {
    unsigned short* C = (unsigned short*)Cout;
#pragma unroll
    for (int t = 0; t < 4; ++t)
#pragma unroll
      for (int v = 0; v < 8; ++v)
        C[(size_t)(m0 + v + h * 8) * ldc + n0 + t * 16 + r] = f2bf(acc[t][v]);
  } else {
    float* C = (float*)Cout;
#pragma unroll
    for (int t = 0; t < 4; ++t)
#pragma unroll
      for (int v = 0; v < 8; ++v)
        C[(size_t)(m0 + v + h * 8) * ldc + n0 + t * 16 + r] = acc[t][v];
  }
}

// ---------------------------------------------------------------------------
// Depthwise causal conv1d (taps=4) + bias + SiLU over the ssm half of xz.
// ---------------------------------------------------------------------------
__global__ __launch_bounds__(256) void conv_silu(
    const unsigned short* __restrict__ xz,   // [NTOK, 2*DINNER] bf16
    const float* __restrict__ w,             // [DINNER, 1, DCONV]
    const float* __restrict__ bias,          // [DINNER]
    unsigned short* __restrict__ xc) {       // [NTOK, DINNER] bf16
  int idx = blockIdx.x * 256 + threadIdx.x;
  if (idx >= NTOK * DINNER) return;
  int d = idx & (DINNER - 1);
  int n = idx >> 10;                         // token index
  int l = n & (SEQ - 1);
  float acc = bias[d];
#pragma unroll
  for (int j = 0; j < DCONV; ++j) {
    int li = l + j - (DCONV - 1);
    if (li >= 0)
      acc += w[d * DCONV + j] * bf2f(xz[(size_t)(n + li - l) * (2 * DINNER) + d]);
  }
  float s = acc / (1.f + __expf(-acc));      // SiLU
  xc[idx] = f2bf(s);
}

// ---------------------------------------------------------------------------
// Selective scan: one thread per (batch, channel). 16 SSM states in regs,
// exp on the trans pipe; fused softplus, gate SiLU, D-skip. Writes y in bf16.
// ---------------------------------------------------------------------------
__global__ __launch_bounds__(256) void selective_scan(
    const float* __restrict__ delta_raw,     // [NTOK, DINNER] (pre-bias)
    const unsigned short* __restrict__ dbc,  // [NTOK, 64] bf16 (dt|B|C)
    const unsigned short* __restrict__ xcb,  // [NTOK, DINNER] bf16
    const unsigned short* __restrict__ xz,   // [NTOK, 2*DINNER] bf16 (z half)
    const float* __restrict__ A_log,         // [DINNER, DSTATE]
    const float* __restrict__ Dvec,          // [DINNER]
    const float* __restrict__ dt_b,          // [DINNER]
    unsigned short* __restrict__ ybf) {      // [NTOK, DINNER] bf16
  const int b  = blockIdx.x >> 2;
  const int d  = ((blockIdx.x & 3) << 8) + threadIdx.x;

  float Arow[DSTATE];
#pragma unroll
  for (int s = 0; s < DSTATE; ++s) Arow[s] = -__expf(A_log[d * DSTATE + s]);
  const float Dd  = Dvec[d];
  const float dtb = dt_b[d];

  float hst[DSTATE];
#pragma unroll
  for (int s = 0; s < DSTATE; ++s) hst[s] = 0.f;

  for (int l = 0; l < SEQ; ++l) {
    const int n = b * SEQ + l;
    float draw  = delta_raw[(size_t)n * DINNER + d] + dtb;
    float delta = (draw > 20.f) ? draw : __logf(1.f + __expf(draw));
    float xc    = bf2f(xcb[(size_t)n * DINNER + d]);
    const unsigned short* bc = dbc + (size_t)n * 64 + DTRANK;   // B then C
    float acc = 0.f;
#pragma unroll
    for (int s = 0; s < DSTATE; ++s) {
      float Bs = bf2f(bc[s]);
      float Cs = bf2f(bc[DSTATE + s]);
      float dA = __expf(delta * Arow[s]);
      hst[s] = dA * hst[s] + delta * Bs * xc;
      acc += hst[s] * Cs;
    }
    float y = acc + Dd * xc;
    float z = bf2f(xz[(size_t)n * (2 * DINNER) + DINNER + d]);
    y *= z / (1.f + __expf(-z));             // y * silu(z)
    ybf[(size_t)n * DINNER + d] = f2bf(y);
  }
}

// ---------------------------------------------------------------------------
// Host-side orchestration
// ---------------------------------------------------------------------------
static constexpr size_t alignup(size_t x) { return (x + 255) & ~size_t(255); }

extern "C" void kernel_launch(void* const* d_in, const int* in_sizes, int n_in,
                              void* d_out, int out_size, void* d_ws, size_t ws_size,
                              hipStream_t stream) {
  const float* x        = (const float*)d_in[0];  // [B,L,512]
  const float* in_w     = (const float*)d_in[1];  // [2048,512]
  const float* conv_w   = (const float*)d_in[2];  // [1024,1,4]
  const float* conv_b   = (const float*)d_in[3];  // [1024]
  const float* xp_w     = (const float*)d_in[4];  // [64,1024]
  const float* dt_w     = (const float*)d_in[5];  // [1024,32]
  const float* dt_b     = (const float*)d_in[6];  // [1024]
  const float* A_log    = (const float*)d_in[7];  // [1024,16]
  const float* Dvec     = (const float*)d_in[8];  // [1024]
  const float* out_w    = (const float*)d_in[9];  // [512,1024]
  float*       out      = (float*)d_out;          // [B,L,512]

  // Workspace layout (bf16 unless noted), 256-B aligned
  char* ws = (char*)d_ws;
  size_t o = 0;
  unsigned short* xbf   = (unsigned short*)(ws + o); o += alignup((size_t)NTOK * DMODEL * 2);
  unsigned short* winb  = (unsigned short*)(ws + o); o += alignup((size_t)2 * DINNER * DMODEL * 2);
  unsigned short* wxpb  = (unsigned short*)(ws + o); o += alignup((size_t)(DTRANK + 2 * DSTATE) * DINNER * 2);
  unsigned short* woutb = (unsigned short*)(ws + o); o += alignup((size_t)DMODEL * DINNER * 2);
  unsigned short* dtwb  = (unsigned short*)(ws + o); o += alignup((size_t)DINNER * DTRANK * 2);
  unsigned short* xzb   = (unsigned short*)(ws + o); o += alignup((size_t)NTOK * 2 * DINNER * 2);
  unsigned short* xcb   = (unsigned short*)(ws + o); o += alignup((size_t)NTOK * DINNER * 2);
  unsigned short* dbcb  = (unsigned short*)(ws + o); o += alignup((size_t)NTOK * 64 * 2);
  float*          draw  = (float*)(ws + o);          o += alignup((size_t)NTOK * DINNER * 4);
  unsigned short* ybf   = (unsigned short*)(ws + o); o += alignup((size_t)NTOK * DINNER * 2);
  (void)ws_size; (void)n_in; (void)in_sizes; (void)out_size;

  auto cvt = [&](const float* src, unsigned short* dst, int n) {
    cvt_f32_bf16<<<(n + 255) / 256, 256, 0, stream>>>(src, dst, n);
  };
  cvt(x,     xbf,   NTOK * DMODEL);
  cvt(in_w,  winb,  2 * DINNER * DMODEL);
  cvt(xp_w,  wxpb,  (DTRANK + 2 * DSTATE) * DINNER);
  cvt(out_w, woutb, DMODEL * DINNER);
  cvt(dt_w,  dtwb,  DINNER * DTRANK);

  auto gemm = [&](const unsigned short* A, int lda, const unsigned short* W, int ldw,
                  void* C, int ldc, int M, int N, int K, int cbf) {
    int waves  = (M >> 4) * (N >> 6);        // 16x64 strip per wave
    int blocks = (waves + 7) / 8;
    gemm_bf16_wmma<<<blocks, 256, 0, stream>>>(A, lda, W, ldw, C, ldc, M, N, K, cbf);
  };

  // 1) xz = x @ in_proj_w^T      [4096, 2048]
  gemm(xbf, DMODEL, winb, DMODEL, xzb, 2 * DINNER, NTOK, 2 * DINNER, DMODEL, 1);

  // 2) causal depthwise conv + SiLU over ssm half -> xc
  conv_silu<<<(NTOK * DINNER + 255) / 256, 256, 0, stream>>>(xzb, conv_w, conv_b, xcb);

  // 3) dbc = xc @ x_proj_w^T     [4096, 64]
  gemm(xcb, DINNER, wxpb, DINNER, dbcb, 64, NTOK, DTRANK + 2 * DSTATE, DINNER, 1);

  // 4) delta_raw = dt @ dt_proj_w^T  (K=32, single WMMA per tile) [4096,1024]
  gemm(dbcb, 64, dtwb, DTRANK, draw, DINNER, NTOK, DINNER, DTRANK, 0);

  // 5) selective scan + gating -> y (bf16)
  selective_scan<<<BATCH * (DINNER / 256), 256, 0, stream>>>(
      draw, dbcb, xcb, xzb, A_log, Dvec, dt_b, ybf);

  // 6) out = y @ out_proj_w^T    [4096, 512] f32 -> d_out
  gemm(ybf, DINNER, woutb, DINNER, out, DMODEL, NTOK, DMODEL, DINNER, 0);
}
